// GraphProp_81492709474574
// MI455X (gfx1250) — matrix operand order repaired and checked by reading, not compiled, not run on hardware
//
#include <hip/hip_runtime.h>

// ---------------------------------------------------------------------------
// GNN message passing on MI455X (gfx1250, wave32):
//   edge MLP (256->256->128, bf16 WMMA, f32 acc) + atomic segment-sum
//   node MLP (256->256->128) + residual
// M_TILE=128 rows/block: one W0+W1 L2 read per 128 rows (weight traffic is the
// dominant L2 term; HBM side is only ~100 MB total).
// ---------------------------------------------------------------------------

typedef __attribute__((ext_vector_type(16))) __bf16 v16bf;
typedef __attribute__((ext_vector_type(8)))  __bf16 v8bf;
typedef __attribute__((ext_vector_type(8)))  float  v8f;

#define D_FEAT   128
#define K_DIM    256   // 2*D concat input width for both MLPs
#define H_DIM    256   // hidden width
#define M_TILE   128   // rows (edges or nodes) per block
#define N_MT     8     // 16-row M-tiles per block (M_TILE/16)
#define THREADS  256   // 8 waves

// f32 -> bf16 round-to-nearest-even, raw u16 (avoids scalar __bf16 arithmetic)
__device__ __forceinline__ unsigned short f2bfu(float f) {
    union { float f; unsigned u; } c; c.f = f;
    unsigned r = c.u + 0x7FFFu + ((c.u >> 16) & 1u);
    return (unsigned short)(r >> 16);
}

// Compute N_MT 16x16 output tiles of column-block n: acc[mt] += A[mt] * Wt[n]^T,
// K = 256 in 8 wmma_bf16 steps. Each B fragment is loaded ONCE per (n, kt) and
// reused across all N_MT M-tiles; A fragments come from LDS (cheap to reload).
__device__ __forceinline__ void mma_col_k256(const unsigned short* A /*[rows][K_DIM]*/,
                                             const unsigned short* Wt /*[N][K_DIM]*/,
                                             int n, int half, int lanlo,
                                             v8f acc[N_MT]) {
#pragma unroll
    for (int kt = 0; kt < 8; ++kt) {
        // B: lanes 0-15 -> K 0..15, lanes 16-31 -> K 16..31 (contiguous in Wt row)
        const int kb = kt * 32 + half * 16;
        const unsigned short* bp = Wt + (size_t)n * K_DIM + kb;
        v8bf blo = *reinterpret_cast<const v8bf*>(bp);
        v8bf bhi = *reinterpret_cast<const v8bf*>(bp + 8);
        v16bf bv;
#pragma unroll
        for (int i = 0; i < 8; ++i) { bv[i] = blo[i]; bv[i + 8] = bhi[i]; }

        const int ka = kt * 32 + half * 16;  // A halves: K 0-15 / 16-31 (contiguous, matches B)
#pragma unroll
        for (int mt = 0; mt < N_MT; ++mt) {
            const unsigned short* ap = A + (size_t)(mt * 16 + lanlo) * K_DIM + ka;
            v8bf alo = *reinterpret_cast<const v8bf*>(ap);
            v8bf ahi = *reinterpret_cast<const v8bf*>(ap + 8);
            v16bf av;
#pragma unroll
            for (int i = 0; i < 8; ++i) { av[i] = alo[i]; av[i + 8] = ahi[i]; }
            acc[mt] = __builtin_amdgcn_wmma_f32_16x16x32_bf16(
                /*neg_a=*/false, av, /*neg_b=*/false, bv,
                /*c_mod=*/(short)0, acc[mt], /*reuse_a=*/false, /*reuse_b=*/false);
        }
    }
}

// ------------------------------- prep kernels ------------------------------

__global__ __launch_bounds__(256) void k_zero_f32(float* p, long n) {
    long i = (long)blockIdx.x * 256 + threadIdx.x;
    if (i < n) p[i] = 0.0f;
}

__global__ __launch_bounds__(256) void k_f32_to_bf16(const float* __restrict__ in,
                                                     unsigned short* __restrict__ out, long n) {
    long i = (long)blockIdx.x * 256 + threadIdx.x;
    if (i < n) out[i] = f2bfu(in[i]);
}

// W is [K][N] row-major f32; Wt is [N][K] bf16 (lane's K-run contiguous)
__global__ __launch_bounds__(256) void k_transpose_bf16(const float* __restrict__ W,
                                                        unsigned short* __restrict__ Wt,
                                                        int K, int N) {
    long i = (long)blockIdx.x * 256 + threadIdx.x;
    if (i < (long)K * N) {
        int k = (int)(i / N), n = (int)(i % N);
        Wt[(size_t)n * K + k] = f2bfu(W[(size_t)k * N + n]);
    }
}

// ------------------------------- edge kernel -------------------------------
// Fused: gather -> [128x256] bf16 tile -> W0(256x256)+b0,relu -> W1(256x128)+b1,relu
//        -> atomic scatter-add into agg[to_idx]
__global__ __launch_bounds__(THREADS) void k_edge_msg(
    const unsigned short* __restrict__ nodeBf,   // [N][128] bf16
    const int* __restrict__ fromIdx, const int* __restrict__ toIdx,
    const unsigned short* __restrict__ w0t,      // [256][256] bf16 (N-major)
    const float* __restrict__ b0,
    const unsigned short* __restrict__ w1t,      // [128][256] bf16
    const float* __restrict__ b1,
    float* __restrict__ agg, int nEdges)
{
    extern __shared__ unsigned short smem[];     // [M_TILE][256] A  +  [M_TILE][256] H
    unsigned short* aLds = smem;                           // 64 KB
    unsigned short* hLds = smem + (size_t)M_TILE * K_DIM;  // 64 KB
    __shared__ int fI[M_TILE], tI[M_TILE];

    const int tid = threadIdx.x;
    const int e0  = blockIdx.x * M_TILE;
    __builtin_prefetch(w0t, 0, 1);               // -> global_prefetch_b8 (L2-hot weights)

    if (tid < M_TILE) {
        int e = e0 + tid; e = (e < nEdges) ? e : 0;
        fI[tid] = fromIdx[e];
        tI[tid] = toIdx[e];
    }
    __syncthreads();

    // Stage A tile: M_TILE rows x 32 chunks of 16B (8 bf16) each
#pragma unroll
    for (int c = tid; c < M_TILE * 32; c += THREADS) {
        const int row = c >> 5, sub = c & 31;
        const int part = sub >> 4;               // 0 = from-feat, 1 = to-feat
        const int off  = (sub & 15) * 8;
        const int node = part ? tI[row] : fI[row];
        const uint4 v = *reinterpret_cast<const uint4*>(nodeBf + (size_t)node * D_FEAT + off);
        *reinterpret_cast<uint4*>(&aLds[(size_t)row * K_DIM + part * D_FEAT + off]) = v;
    }
    __syncthreads();

    const int wave = tid >> 5, lane = tid & 31;
    const int half = lane >> 4, lanlo = lane & 15;

    // Layer 0: 16 N-tiles x 8 M-tiles; each wave owns 2 N-tiles (sequential passes)
    for (int nti = 0; nti < 2; ++nti) {
        const int n = (wave * 2 + nti) * 16 + lanlo;
        const float bias = b0[n];
        v8f acc[N_MT] = {};
        mma_col_k256(aLds, w0t, n, half, lanlo, acc);
#pragma unroll
        for (int mt = 0; mt < N_MT; ++mt)
#pragma unroll
            for (int r = 0; r < 8; ++r) {
                const int m = mt * 16 + half * 8 + r;   // C/D: VGPR r -> M = r + 8*(lane/16)
                float v = acc[mt][r] + bias; v = v > 0.0f ? v : 0.0f;
                hLds[(size_t)m * H_DIM + n] = f2bfu(v);
            }
    }
    __syncthreads();

    // Layer 1: N = 128 -> 8 N-tiles, one per wave; fused ReLU + atomic scatter
    {
        const int n = wave * 16 + lanlo;
        const float bias = b1[n];
        v8f acc[N_MT] = {};
        mma_col_k256(hLds, w1t, n, half, lanlo, acc);
#pragma unroll
        for (int mt = 0; mt < N_MT; ++mt)
#pragma unroll
            for (int r = 0; r < 8; ++r) {
                const int m = mt * 16 + half * 8 + r;
                if (e0 + m < nEdges) {
                    float v = acc[mt][r] + bias; v = v > 0.0f ? v : 0.0f;
                    atomicAdd(&agg[(size_t)tI[m] * D_FEAT + n], v);
                }
            }
    }
}

// ------------------------------- node kernel -------------------------------
// Fused: [agg||feat] -> W0+b0,relu -> W1+b1,relu -> residual add -> out (f32)
__global__ __launch_bounds__(THREADS) void k_node_upd(
    const float* __restrict__ nodeF,             // [N][128] f32
    const unsigned short* __restrict__ nodeBf,   // [N][128] bf16
    const float* __restrict__ agg,               // [N][128] f32
    const unsigned short* __restrict__ w0t, const float* __restrict__ b0,
    const unsigned short* __restrict__ w1t, const float* __restrict__ b1,
    float* __restrict__ out, int nNodes)
{
    extern __shared__ unsigned short smem[];
    unsigned short* aLds = smem;
    unsigned short* hLds = smem + (size_t)M_TILE * K_DIM;

    const int tid = threadIdx.x;
    const int v0  = blockIdx.x * M_TILE;
    __builtin_prefetch(w0t, 0, 1);

    // Stage A tile: per row, cols 0..127 = bf16(agg) (32 float4 chunks),
    //               cols 128..255 = nodeBf (16 uint4 chunks of 8 bf16)
#pragma unroll
    for (int c = tid; c < M_TILE * 48; c += THREADS) {
        const int row = c / 48, s = c % 48;
        int node = v0 + row; node = (node < nNodes) ? node : 0;
        if (s < 32) {                     // agg: convert 4 f32 -> 4 bf16
            const float4 v = *reinterpret_cast<const float4*>(agg + (size_t)node * D_FEAT + s * 4);
            ushort4 o;
            o.x = f2bfu(v.x); o.y = f2bfu(v.y); o.z = f2bfu(v.z); o.w = f2bfu(v.w);
            *reinterpret_cast<ushort4*>(&aLds[(size_t)row * K_DIM + s * 4]) = o;
        } else {                          // features: copy 8 bf16
            const int j = s - 32;
            const uint4 v = *reinterpret_cast<const uint4*>(nodeBf + (size_t)node * D_FEAT + j * 8);
            *reinterpret_cast<uint4*>(&aLds[(size_t)row * K_DIM + D_FEAT + j * 8]) = v;
        }
    }
    __syncthreads();

    const int wave = tid >> 5, lane = tid & 31;
    const int half = lane >> 4, lanlo = lane & 15;

    for (int nti = 0; nti < 2; ++nti) {
        const int n = (wave * 2 + nti) * 16 + lanlo;
        const float bias = b0[n];
        v8f acc[N_MT] = {};
        mma_col_k256(aLds, w0t, n, half, lanlo, acc);
#pragma unroll
        for (int mt = 0; mt < N_MT; ++mt)
#pragma unroll
            for (int r = 0; r < 8; ++r) {
                const int m = mt * 16 + half * 8 + r;
                float v = acc[mt][r] + bias; v = v > 0.0f ? v : 0.0f;
                hLds[(size_t)m * H_DIM + n] = f2bfu(v);
            }
    }
    __syncthreads();

    {
        const int n = wave * 16 + lanlo;
        const float bias = b1[n];
        v8f acc[N_MT] = {};
        mma_col_k256(hLds, w1t, n, half, lanlo, acc);
#pragma unroll
        for (int mt = 0; mt < N_MT; ++mt)
#pragma unroll
            for (int r = 0; r < 8; ++r) {
                const int m = mt * 16 + half * 8 + r;
                const int node = v0 + m;
                if (node < nNodes) {
                    float v = acc[mt][r] + bias; v = v > 0.0f ? v : 0.0f;
                    out[(size_t)node * D_FEAT + n] = nodeF[(size_t)node * D_FEAT + n] + v;
                }
            }
    }
}

// ---------------------------------- host -----------------------------------

extern "C" void kernel_launch(void* const* d_in, const int* in_sizes, int n_in,
                              void* d_out, int out_size, void* d_ws, size_t ws_size,
                              hipStream_t stream) {
    const float* nodeF  = (const float*)d_in[0];
    const int*   fromIx = (const int*)  d_in[1];
    const int*   toIx   = (const int*)  d_in[2];
    const float* msgW0  = (const float*)d_in[3];
    const float* msgB0  = (const float*)d_in[4];
    const float* msgW1  = (const float*)d_in[5];
    const float* msgB1  = (const float*)d_in[6];
    const float* updW0  = (const float*)d_in[7];
    const float* updB0  = (const float*)d_in[8];
    const float* updW1  = (const float*)d_in[9];
    const float* updB1  = (const float*)d_in[10];

    const int nNodes = in_sizes[0] / D_FEAT;
    const int nEdges = in_sizes[1];

    // Workspace carve-up (256B aligned)
    char* ws = (char*)d_ws;
    size_t off = 0;
    auto carve = [&](size_t bytes) -> void* {
        void* p = ws + off;
        off = (off + bytes + 255) & ~(size_t)255;
        return p;
    };
    float*          agg    = (float*)         carve((size_t)nNodes * D_FEAT * sizeof(float));
    unsigned short* nodeBf = (unsigned short*)carve((size_t)nNodes * D_FEAT * sizeof(unsigned short));
    unsigned short* mW0t   = (unsigned short*)carve((size_t)H_DIM * K_DIM * sizeof(unsigned short));
    unsigned short* mW1t   = (unsigned short*)carve((size_t)D_FEAT * K_DIM * sizeof(unsigned short));
    unsigned short* uW0t   = (unsigned short*)carve((size_t)H_DIM * K_DIM * sizeof(unsigned short));
    unsigned short* uW1t   = (unsigned short*)carve((size_t)D_FEAT * K_DIM * sizeof(unsigned short));
    (void)ws_size; (void)n_in; (void)out_size;

    const long nFeatElems = (long)nNodes * D_FEAT;
    const int  featBlocks = (int)((nFeatElems + 255) / 256);

    // Prep: zero aggregation buffer, bf16 feature table, transposed bf16 weights
    k_zero_f32   <<<featBlocks, 256, 0, stream>>>(agg, nFeatElems);
    k_f32_to_bf16<<<featBlocks, 256, 0, stream>>>(nodeF, nodeBf, nFeatElems);
    k_transpose_bf16<<<(K_DIM * H_DIM  + 255) / 256, 256, 0, stream>>>(msgW0, mW0t, K_DIM, H_DIM);
    k_transpose_bf16<<<(K_DIM * D_FEAT + 255) / 256, 256, 0, stream>>>(msgW1, mW1t, K_DIM, D_FEAT);
    k_transpose_bf16<<<(K_DIM * H_DIM  + 255) / 256, 256, 0, stream>>>(updW0, uW0t, K_DIM, H_DIM);
    k_transpose_bf16<<<(K_DIM * D_FEAT + 255) / 256, 256, 0, stream>>>(updW1, uW1t, K_DIM, D_FEAT);

    const size_t ldsBytes = (size_t)2 * M_TILE * K_DIM * sizeof(unsigned short); // 128 KB

    // Edge message MLP + scatter-add (WMMA bf16, f32 atomics into L2-resident agg)
    const int edgeBlocks = (nEdges + M_TILE - 1) / M_TILE;
    k_edge_msg<<<edgeBlocks, THREADS, ldsBytes, stream>>>(nodeBf, fromIx, toIx,
                                                          mW0t, msgB0, mW1t, msgB1,
                                                          agg, nEdges);

    // Node update MLP + residual
    const int nodeBlocks = (nNodes + M_TILE - 1) / M_TILE;
    k_node_upd<<<nodeBlocks, THREADS, ldsBytes, stream>>>(nodeF, nodeBf, agg,
                                                          uW0t, updB0, uW1t, updB1,
                                                          (float*)d_out, nNodes);
}
